// GSC_RecurrentSNN_15831249453630
// MI455X (gfx1250) — matrix operand order repaired and checked by reading, compiled, not run on hardware
//
#include <hip/hip_runtime.h>

#define B_SZ    1024
#define T_STEPS  100
#define N_INPUT  120
#define K1P      128   // x K padded to multiple of 32
#define H1       512
#define H2       256
#define N_OUT     35
#define W3R       48   // W3 rows padded to 3 col-tiles of 16
#define APITCH    40   // LDS A-tile pitch (bf16 elems); 80B rows dodge bank conflicts

typedef __bf16 v16bf __attribute__((ext_vector_type(16)));
typedef float  v8f   __attribute__((ext_vector_type(8)));

__device__ __forceinline__ __bf16 f2bf(float f) {
  union { float f; unsigned u; } x; x.f = f;
  union { unsigned short u; __bf16 b; } r;
  r.u = (unsigned short)((x.u + 0x7FFFu + ((x.u >> 16) & 1u)) >> 16); // RNE
  return r.b;
}
__device__ __forceinline__ float bf2f(__bf16 b) {
  union { __bf16 b; unsigned short u; } t; t.b = b;
  union { unsigned u; float f; } r; r.u = ((unsigned)t.u) << 16;
  return r.f;
}

__device__ __forceinline__ v8f wmma_bf16(v16bf a, v16bf b, v8f c) {
  return __builtin_amdgcn_wmma_f32_16x16x32_bf16(false, a, false, b, (short)0, c,
                                                 false, false);
}

// A fragment (16x32 bf16) from an LDS tile [16][APITCH].
// Lanes 0-15: M=lane, K={0..7,16..23}; lanes 16-31: M=lane-16, K={8..15,24..31}.
__device__ __forceinline__ v16bf load_a_lds(const __bf16* a_lds, int lane) {
  const __bf16* p = a_lds + (lane & 15) * APITCH + ((lane & 16) ? 8 : 0);
  v16bf f;
#pragma unroll
  for (int j = 0; j < 8; ++j) { f[j] = p[j]; f[8 + j] = p[16 + j]; }
  return f;
}

// A fragment straight from a global bf16 matrix [M][kpitch].
__device__ __forceinline__ v16bf load_a_g(const __bf16* a, int m0, int k0, int kpitch,
                                          int lane) {
  const __bf16* p = a + (size_t)(m0 + (lane & 15)) * kpitch + k0 + ((lane & 16) ? 8 : 0);
  v16bf f;
#pragma unroll
  for (int j = 0; j < 8; ++j) { f[j] = p[j]; f[8 + j] = p[16 + j]; }
  return f;
}

// B fragment (32x16): weights stored [N][kpitch] (B[k][n] = W[n][k]).
// Lanes 0-15: N=n0+lane, K=k0+{0..7,16..23}; lanes 16-31: N=n0+lane-16, K=k0+{8..15,24..31}.
__device__ __forceinline__ v16bf load_b(const __bf16* w, int n0, int k0, int kpitch,
                                        int lane) {
  const __bf16* p = w + (size_t)(n0 + (lane & 15)) * kpitch + k0 + ((lane & 16) ? 8 : 0);
  v16bf f;
#pragma unroll
  for (int j = 0; j < 8; ++j) { f[j] = p[j]; f[8 + j] = p[16 + j]; }
  return f;
}

// Cooperative 16x32 bf16 tile fill: 64 threads x 16B (global_b128 -> ds_store_b128).
__device__ __forceinline__ void fill_tile(__bf16* dst, const __bf16* srcRow0,
                                          long row_stride, int k0) {
  const int tid = threadIdx.x;
  if (tid < 64) {
    const int r = tid >> 2;
    const int c = (tid & 3) * 8;
    *(uint4*)(dst + r * APITCH + c) =
        *(const uint4*)(srcRow0 + (long)r * row_stride + k0 + c);
  }
}

// ---------------- init: bf16 weights + padded bf16 x, zero all state -----------
__global__ void snn_init(const float* __restrict__ x, const float* __restrict__ W1,
                         const float* __restrict__ Wrec, const float* __restrict__ W2,
                         const float* __restrict__ W3, __bf16* xb16, __bf16* w1b,
                         __bf16* wrb, __bf16* w2b, __bf16* w3b, float* v1, float* a1,
                         float* v2, float* a2, __bf16* s1, __bf16* s2, __bf16* s1f,
                         __bf16* s2f, float* vo, float* out) {
  const long stride = (long)gridDim.x * blockDim.x;
  const long tid = (long)blockIdx.x * blockDim.x + threadIdx.x;
  const __bf16 z = f2bf(0.f);
  for (long i = tid; i < (long)B_SZ * T_STEPS * K1P; i += stride) {
    long bt = i >> 7; int k = (int)(i & 127);
    xb16[i] = (k < N_INPUT) ? f2bf(x[bt * N_INPUT + k]) : z;
  }
  for (long i = tid; i < H1 * K1P; i += stride) {
    int n = (int)(i >> 7), k = (int)(i & 127);
    w1b[i] = (k < N_INPUT) ? f2bf(W1[n * N_INPUT + k]) : z;
  }
  for (long i = tid; i < H1 * H1; i += stride) wrb[i] = f2bf(Wrec[i]);
  for (long i = tid; i < H2 * H1; i += stride) w2b[i] = f2bf(W2[i]);
  for (long i = tid; i < W3R * H2; i += stride)
    w3b[i] = (i < N_OUT * H2) ? f2bf(W3[i]) : z;
  for (long i = tid; i < (long)B_SZ * H1; i += stride) {
    v1[i] = 0.f; a1[i] = 0.f;
    s1[i] = z; s1[(long)B_SZ * H1 + i] = z;
    s1f[i] = z; s1f[(long)B_SZ * H1 + i] = z;
  }
  for (long i = tid; i < (long)B_SZ * H2; i += stride) {
    v2[i] = 0.f; a2[i] = 0.f;
    s2[i] = z; s2[(long)B_SZ * H2 + i] = z;
    s2f[i] = z; s2f[(long)B_SZ * H2 + i] = z;
  }
  for (long i = tid; i < B_SZ * N_OUT; i += stride) { vo[i] = 0.f; out[i] = 0.f; }
}

// ------------- fused GEMM (optional A0|A1 K-concat) + adaptive LIF -------------
// Block: 256 thr = 8 waves; wave computes NACC 16x16 C tiles sharing one A frag.
// Double-buffered LDS A tile: one barrier per 32-wide K step; two split K loops
// (x phase, recurrent phase) avoid per-iteration operand selects.
// v/a/spike-shadow state in per-thread-contiguous fragment layout:
//   index = (tile*32 + lane)*8 + vgpr  -> 32B/thread runs, fully coalesced.
// s_next additionally scattered to standard [B][ncols] for the next GEMM's A.
template <int NACC>
__global__ void __launch_bounds__(256) snn_gemm_lif(
    const __bf16* __restrict__ A0, long a0_stride, int i0cnt,
    const __bf16* __restrict__ W0, const __bf16* __restrict__ A1, int K1,
    const __bf16* __restrict__ W1w, int ncols, int col_groups,
    const __bf16* __restrict__ spf_prev, __bf16* __restrict__ spf_next,
    __bf16* __restrict__ s_next, float* __restrict__ v, float* __restrict__ a,
    const float* __restrict__ alpha, const float* __restrict__ rho,
    const float* __restrict__ beta_a) {
  __shared__ __bf16 lds[2][16 * APITCH];
  const int lane = threadIdx.x & 31;
  const int wave = threadIdx.x >> 5;
  const int row0 = (blockIdx.x / col_groups) * 16;
  const int colw = (blockIdx.x % col_groups) * (8 * NACC * 16) + wave * (NACC * 16);
  const int K0 = i0cnt * 32;
  const int nIter = i0cnt + (K1 >> 5);

  const __bf16* a0row = A0 + (long)row0 * a0_stride;
  const __bf16* a1row = A1 + (long)row0 * K1;

  auto fill_idx = [&](int idx, int buf) {
    if (idx < i0cnt) fill_tile(lds[buf], a0row, a0_stride, idx * 32);
    else             fill_tile(lds[buf], a1row, K1, (idx - i0cnt) * 32);
  };

  fill_idx(0, 0);
  __syncthreads();

  v8f acc[NACC] = {};
  int it = 0;
  for (; it < i0cnt; ++it) {               // x-input phase
    const int cur = it & 1;
    if (it + 1 < nIter) fill_idx(it + 1, cur ^ 1);
    const v16bf afr = load_a_lds(lds[cur], lane);
#pragma unroll
    for (int j = 0; j < NACC; ++j)
      acc[j] = wmma_bf16(afr, load_b(W0, colw + j * 16, it * 32, K0, lane), acc[j]);
    __syncthreads();
  }
  for (; it < nIter; ++it) {               // recurrent/feed-forward phase
    const int cur = it & 1;
    if (it + 1 < nIter) fill_idx(it + 1, cur ^ 1);
    const v16bf afr = load_a_lds(lds[cur], lane);
    const int k0 = (it - i0cnt) * 32;
#pragma unroll
    for (int j = 0; j < NACC; ++j)
      acc[j] = wmma_bf16(afr, load_b(W1w, colw + j * 16, k0, K1, lane), acc[j]);
    __syncthreads();
  }

  // adaptive-LIF epilogue; C layout: VGPR i -> M = i + 8*(lane>=16), N = lane&15
  const int mbase = row0 + ((lane & 16) ? 8 : 0);
#pragma unroll
  for (int j = 0; j < NACC; ++j) {
    const int n = colw + j * 16 + (lane & 15);
    const float al = alpha[n], rh = rho[n], ba = beta_a[n];
    const size_t tb =
        (((size_t)(row0 >> 4) * (ncols >> 4) + ((colw + j * 16) >> 4)) * 32 + lane) * 8;
#pragma unroll
    for (int i = 0; i < 8; ++i) {
      const size_t fo = tb + i;                        // contiguous per thread
      const float sp = bf2f(spf_prev[fo]);
      const float an = rh * a[fo] + (1.f - rh) * sp;
      const float vn = al * v[fo] * (1.f - sp) + (1.f - al) * acc[j][i];
      const float sn = (vn > 1.f + ba * an) ? 1.f : 0.f;
      v[fo] = vn; a[fo] = an;
      const __bf16 sb = f2bf(sn);
      spf_next[fo] = sb;
      s_next[(size_t)(mbase + i) * ncols + n] = sb;    // standard layout for GEMM-A
    }
  }
}

// ---------------- output: vo = b*vo + (1-b)*(s2@W3^T); out += vo/T -------------
// 8 blocks x 8 waves = 64 waves; each wave: 16 rows x all 3 col tiles (48 padded).
__global__ void __launch_bounds__(256) snn_out(
    const __bf16* __restrict__ s2, const __bf16* __restrict__ w3b,
    const float* __restrict__ beta_out, float* __restrict__ vo,
    float* __restrict__ out) {
  const int lane = threadIdx.x & 31;
  const int row0 = (blockIdx.x * 8 + (threadIdx.x >> 5)) * 16;
  v8f acc[3] = {};
#pragma unroll
  for (int k0 = 0; k0 < H2; k0 += 32) {
    const v16bf afr = load_a_g(s2, row0, k0, H2, lane);
#pragma unroll
    for (int j = 0; j < 3; ++j)
      acc[j] = wmma_bf16(afr, load_b(w3b, j * 16, k0, H2, lane), acc[j]);
  }
  const int mbase = row0 + ((lane & 16) ? 8 : 0);
#pragma unroll
  for (int j = 0; j < 3; ++j) {
    const int n = j * 16 + (lane & 15);
    if (n < N_OUT) {
      const float bo = beta_out[n];
#pragma unroll
      for (int i = 0; i < 8; ++i) {
        const size_t off = (size_t)(mbase + i) * N_OUT + n;
        const float vn = bo * vo[off] + (1.f - bo) * acc[j][i];
        vo[off] = vn;
        out[off] += vn * (1.f / T_STEPS);
      }
    }
  }
}

extern "C" void kernel_launch(void* const* d_in, const int* in_sizes, int n_in,
                              void* d_out, int out_size, void* d_ws, size_t ws_size,
                              hipStream_t stream) {
  (void)in_sizes; (void)n_in; (void)out_size; (void)ws_size;
  const float* x       = (const float*)d_in[0];
  const float* W1      = (const float*)d_in[1];
  const float* Wrec    = (const float*)d_in[2];
  const float* W2      = (const float*)d_in[3];
  const float* W3      = (const float*)d_in[4];
  const float* alpha1  = (const float*)d_in[5];
  const float* rho1    = (const float*)d_in[6];
  const float* beta_a1 = (const float*)d_in[7];
  const float* alpha2  = (const float*)d_in[8];
  const float* rho2    = (const float*)d_in[9];
  const float* beta_a2 = (const float*)d_in[10];
  const float* beta_o  = (const float*)d_in[11];
  float* out = (float*)d_out;

  char* p = (char*)d_ws;
  auto take = [&](size_t bytes) -> char* {
    char* r = p; p += (bytes + 255) & ~(size_t)255; return r;
  };
  __bf16* xb16 = (__bf16*)take((size_t)B_SZ * T_STEPS * K1P * 2);
  __bf16* w1b  = (__bf16*)take((size_t)H1 * K1P * 2);
  __bf16* wrb  = (__bf16*)take((size_t)H1 * H1 * 2);
  __bf16* w2b  = (__bf16*)take((size_t)H2 * H1 * 2);
  __bf16* w3b  = (__bf16*)take((size_t)W3R * H2 * 2);
  __bf16* s1   = (__bf16*)take((size_t)2 * B_SZ * H1 * 2);  // ping-pong, standard
  __bf16* s2   = (__bf16*)take((size_t)2 * B_SZ * H2 * 2);
  __bf16* s1f  = (__bf16*)take((size_t)2 * B_SZ * H1 * 2);  // ping-pong, frag layout
  __bf16* s2f  = (__bf16*)take((size_t)2 * B_SZ * H2 * 2);
  float*  v1   = (float*)take((size_t)B_SZ * H1 * 4);       // frag layout
  float*  a1   = (float*)take((size_t)B_SZ * H1 * 4);
  float*  v2   = (float*)take((size_t)B_SZ * H2 * 4);
  float*  a2   = (float*)take((size_t)B_SZ * H2 * 4);
  float*  vo   = (float*)take((size_t)B_SZ * N_OUT * 4);

  snn_init<<<1024, 256, 0, stream>>>(x, W1, Wrec, W2, W3, xb16, w1b, wrb, w2b, w3b,
                                     v1, a1, v2, a2, s1, s2, s1f, s2f, vo, out);

  for (int t = 0; t < T_STEPS; ++t) {
    const size_t pp = (size_t)(t & 1);
    __bf16* s1p  = s1  + pp * B_SZ * H1;
    __bf16* s1n  = s1  + (pp ^ 1) * B_SZ * H1;
    __bf16* s1fp = s1f + pp * B_SZ * H1;
    __bf16* s1fn = s1f + (pp ^ 1) * B_SZ * H1;
    __bf16* s2p  = s2  + pp * B_SZ * H2;
    __bf16* s2n  = s2  + (pp ^ 1) * B_SZ * H2;
    __bf16* s2fp = s2f + pp * B_SZ * H2;
    __bf16* s2fn = s2f + (pp ^ 1) * B_SZ * H2;
    // layer 1: A-K = [x_pad(128) | s1(512)], 512 cols -> 2 col groups, grid 128
    snn_gemm_lif<2><<<128, 256, 0, stream>>>(
        xb16 + (size_t)t * K1P, (long)T_STEPS * K1P, K1P / 32, w1b, s1p, H1, wrb, H1,
        2, s1fp, s1fn, s1n, v1, a1, alpha1, rho1, beta_a1);
    // layer 2: A-K = s1(512), 256 cols -> 1 col group, grid 64
    snn_gemm_lif<2><<<64, 256, 0, stream>>>(
        nullptr, 0, 0, nullptr, s1n, H1, w2b, H2, 1, s2fp, s2fn, s2n, v2, a2, alpha2,
        rho2, beta_a2);
    snn_out<<<8, 256, 0, stream>>>(s2n, w3b, beta_o, vo, out);
  }
}
